// NetGCN_57844619542974
// MI455X (gfx1250) — matrix operand order repaired and checked by reading, not compiled, not run on hardware
//
#include <hip/hip_runtime.h>

typedef __attribute__((ext_vector_type(16))) _Float16 v16h;
typedef __attribute__((ext_vector_type(8)))  float    v8f;
typedef __attribute__((ext_vector_type(2)))  float    v2f;

#define WMMA_F16(a,b,c) \
  __builtin_amdgcn_wmma_f32_16x16x32_f16(false,(a),false,(b),(short)0,(c),false,false)

// K-offset pattern for 16-bit A-matrix 16x32 (CDNA5 ISA 7.12.2):
// VGPR v<4: K = 2v + 8*hi ; VGPR v>=4: K = 16 + 2(v-4) + 8*hi   (two halves per VGPR)
__device__ __forceinline__ int kpat(int v, int hi) {
  return (v < 4) ? (2*v + 8*hi) : (16 + 2*(v-4) + 8*hi);
}

// Load one A-fragment (16x32 f16) for this lane from row-major f32.
// p = &A[row*ldK + kt*32]; TF: fused "relu then per-K affine" (BN folded), branch-free.
template <bool TF>
__device__ __forceinline__ v16h load_a_g(const float* __restrict__ p, int hi,
                                         const float* __restrict__ tsc,
                                         const float* __restrict__ tsh, int kbase) {
  v16h a;
#pragma unroll
  for (int v = 0; v < 8; ++v) {
    int k = kpat(v, hi);
    v2f x = *(const v2f*)(p + k);       // k is even -> 8B aligned
    float x0 = x.x, x1 = x.y;
    if (TF) {
      int kk = kbase + k;
      x0 = fmaxf(x0, 0.f) * tsc[kk]   + tsh[kk];
      x1 = fmaxf(x1, 0.f) * tsc[kk+1] + tsh[kk+1];
    }
    a[2*v] = (_Float16)x0; a[2*v+1] = (_Float16)x1;
  }
  return a;
}

// Pre-packed B-fragment: 32 contiguous bytes per (kt,nt,lane) -> 2x b128 loads.
__device__ __forceinline__ v16h load_b_p(const _Float16* __restrict__ Wp,
                                         int kt, int nt, int lane) {
  return *(const v16h*)(Wp + ((((size_t)kt*4 + nt)*32 + lane) << 4));
}

// Pack W[K][64] (row-major f32) into fragment-ordered f16:
// dst[kt*2048 + nt*512 + lane*16 + j] = (f16) W[kt*32 + (j<8 ? 8*(lane>>4) + j : 16 + 8*(lane>>4) + j-8)][nt*16 + (lane&15)]
__global__ void k_packW(const float* __restrict__ W, int K, _Float16* __restrict__ dst) {
  int idx = blockIdx.x * blockDim.x + threadIdx.x;
  int total = (K >> 5) * 2048;
  if (idx >= total) return;
  int j    = idx & 15;
  int lane = (idx >> 4) & 31;
  int nt   = (idx >> 9) & 3;
  int kt   = idx >> 11;
  int hi = lane >> 4, nlo = lane & 15;
  int k = kt*32 + (j < 8 ? 8*hi + j : 16 + 8*hi + (j - 8));
  int c = nt*16 + nlo;
  dst[idx] = (_Float16)W[(size_t)k * 64 + c];
}

// ---------------- generic GEMM: out[M,64] = tf(A[M,K]) @ W[K,64] + bias ----------------
template <bool TF>
__global__ void __launch_bounds__(32)
k_gemm(const float* __restrict__ A, int K, long M,
       const _Float16* __restrict__ Wp, const float* __restrict__ bias,
       const float* __restrict__ tsc, const float* __restrict__ tsh,
       float* __restrict__ out) {
  const int lane = threadIdx.x & 31, hi = lane >> 4, nlo = lane & 15;
  const long r0 = (long)blockIdx.x * 16;
  long arow = r0 + nlo; if (arow > M - 1) arow = M - 1;
  v8f acc[4] = {};
  const int KT = K >> 5;
  for (int kt = 0; kt < KT; ++kt) {
    v16h a = load_a_g<TF>(A + arow * K + kt*32, hi, tsc, tsh, kt*32);
#pragma unroll
    for (int nt = 0; nt < 4; ++nt)
      acc[nt] = WMMA_F16(a, load_b_p(Wp, kt, nt, lane), acc[nt]);
  }
#pragma unroll
  for (int nt = 0; nt < 4; ++nt) {
    const int c = nt*16 + nlo;
    const float bs = bias[c];
#pragma unroll
    for (int r = 0; r < 8; ++r) {
      long m = r0 + r + 8*hi;
      if (m < M) out[m*64 + c] = acc[nt][r] + bs;
    }
  }
}

// -------- message pass: e = EA@We+be ; scatter-add norm*relu(xl[row]+e) into hout[col] ----
__global__ void __launch_bounds__(32)
k_msg(const float* __restrict__ EA, const _Float16* __restrict__ Wep,
      const float* __restrict__ be,
      const int* __restrict__ ei, long E,
      const float* __restrict__ xl, const float* __restrict__ dinv,
      float* __restrict__ hout) {
  const int lane = threadIdx.x & 31, hi = lane >> 4, nlo = lane & 15;
  const long e0 = (long)blockIdx.x * 16;
  long ar = e0 + nlo; if (ar > E - 1) ar = E - 1;
  v8f acc[4] = {};
#pragma unroll
  for (int kt = 0; kt < 2; ++kt) {
    v16h a = load_a_g<false>(EA + ar*64 + kt*32, hi, nullptr, nullptr, 0);
#pragma unroll
    for (int nt = 0; nt < 4; ++nt)
      acc[nt] = WMMA_F16(a, load_b_p(Wep, kt, nt, lane), acc[nt]);
  }
  const int* rowp = ei; const int* colp = ei + E;
#pragma unroll
  for (int r = 0; r < 8; ++r) {
    long m = e0 + r + 8*hi;
    if (m >= E) continue;
    int ri = rowp[m], ci = colp[m];
    float nrm = dinv[ri] * dinv[ci];
    const float* xr = xl + (long)ri * 64;
    float* ho = hout + (long)ci * 64;
#pragma unroll
    for (int nt = 0; nt < 4; ++nt) {
      int c = nt*16 + nlo;
      float v = fmaxf(acc[nt][r] + be[c] + xr[c], 0.f) * nrm;
      atomicAdd(&ho[c], v);
    }
  }
}

// -------- new_e tile accumulation: acc = EA@W3e + (h3[row]+h3[col])@Wu (biases in epilogue) --
__device__ __forceinline__ void newe_acc(v8f acc[4],
    const float* __restrict__ EA, const _Float16* __restrict__ W3ep,
    const _Float16* __restrict__ Wup,
    const int* __restrict__ ei, long E, const float* __restrict__ h3,
    long e0, int lane, int hi, int nlo) {
  long ar = e0 + nlo; if (ar > E - 1) ar = E - 1;
#pragma unroll
  for (int kt = 0; kt < 2; ++kt) {
    v16h a = load_a_g<false>(EA + ar*64 + kt*32, hi, nullptr, nullptr, 0);
#pragma unroll
    for (int nt = 0; nt < 4; ++nt)
      acc[nt] = WMMA_F16(a, load_b_p(W3ep, kt, nt, lane), acc[nt]);
  }
  int ri = ei[ar], ci = ei[E + ar];
  const float* pr = h3 + (long)ri * 64;
  const float* pc = h3 + (long)ci * 64;
#pragma unroll
  for (int kt = 0; kt < 2; ++kt) {
    v16h a;
#pragma unroll
    for (int v = 0; v < 8; ++v) {
      int k = kt*32 + kpat(v, hi);
      a[2*v]   = (_Float16)(pr[k]   + pc[k]);
      a[2*v+1] = (_Float16)(pr[k+1] + pc[k+1]);
    }
#pragma unroll
    for (int nt = 0; nt < 4; ++nt)
      acc[nt] = WMMA_F16(a, load_b_p(Wup, kt, nt, lane), acc[nt]);
  }
}

// pass A: accumulate per-column sum/sumsq of relu(new_e) for edge BN
__global__ void __launch_bounds__(32)
k_newe_stats(const float* __restrict__ EA, const _Float16* __restrict__ W3ep,
             const float* __restrict__ b3e,
             const _Float16* __restrict__ Wup, const float* __restrict__ bu,
             const int* __restrict__ ei, long E, const float* __restrict__ h3,
             float* __restrict__ stats) {
  const int lane = threadIdx.x & 31, hi = lane >> 4, nlo = lane & 15;
  const long e0 = (long)blockIdx.x * 16;
  v8f acc[4] = {};
  newe_acc(acc, EA, W3ep, Wup, ei, E, h3, e0, lane, hi, nlo);
#pragma unroll
  for (int nt = 0; nt < 4; ++nt) {
    int c = nt*16 + nlo; float bb = bu[c] + b3e[c];
    float s = 0.f, q = 0.f;
#pragma unroll
    for (int r = 0; r < 8; ++r) {
      long m = e0 + r + 8*hi; if (m >= E) continue;
      float v = fmaxf(acc[nt][r] + bb, 0.f);
      s += v; q += v * v;
    }
    atomicAdd(&stats[c], s); atomicAdd(&stats[64 + c], q);
  }
}

// pass B: recompute new_e, apply BN(relu), LDS reshape C->A, @We_lin + be_lin -> edge_out
__global__ void __launch_bounds__(32)
k_edge_out(const float* __restrict__ EA, const _Float16* __restrict__ W3ep,
           const float* __restrict__ b3e,
           const _Float16* __restrict__ Wup, const float* __restrict__ bu,
           const int* __restrict__ ei, long E, const float* __restrict__ h3,
           const float* __restrict__ tsc, const float* __restrict__ tsh,
           const _Float16* __restrict__ Wlp, const float* __restrict__ bl,
           float* __restrict__ out) {
  __shared__ _Float16 sA[16 * 64];
  const int lane = threadIdx.x & 31, hi = lane >> 4, nlo = lane & 15;
  const long e0 = (long)blockIdx.x * 16;
  v8f acc[4] = {};
  newe_acc(acc, EA, W3ep, Wup, ei, E, h3, e0, lane, hi, nlo);
#pragma unroll
  for (int nt = 0; nt < 4; ++nt) {
    int c = nt*16 + nlo; float bb = bu[c] + b3e[c];
#pragma unroll
    for (int r = 0; r < 8; ++r) {
      float v = fmaxf(acc[nt][r] + bb, 0.f) * tsc[c] + tsh[c];
      sA[(r + 8*hi) * 64 + c] = (_Float16)v;
    }
  }
  __syncthreads();
  v8f a2[4] = {};
#pragma unroll
  for (int kt = 0; kt < 2; ++kt) {
    v16h a;
#pragma unroll
    for (int v = 0; v < 8; ++v) {
      int k = kt*32 + kpat(v, hi);
      a[2*v] = sA[nlo*64 + k]; a[2*v+1] = sA[nlo*64 + k + 1];
    }
#pragma unroll
    for (int nt = 0; nt < 4; ++nt)
      a2[nt] = WMMA_F16(a, load_b_p(Wlp, kt, nt, lane), a2[nt]);
  }
#pragma unroll
  for (int nt = 0; nt < 4; ++nt) {
    int c = nt*16 + nlo; float bs = bl[c];
#pragma unroll
    for (int r = 0; r < 8; ++r) {
      long m = e0 + r + 8*hi;
      if (m < E) out[m*64 + c] = a2[nt][r] + bs;
    }
  }
}

// ---------------- small helpers ----------------
__global__ void k_zero(float* __restrict__ p, long n) {
  long i = (long)blockIdx.x * blockDim.x + threadIdx.x;
  if (i < n) p[i] = 0.f;
}
__global__ void k_deg(const int* __restrict__ row, float* __restrict__ deg, long E) {
  long i = (long)blockIdx.x * blockDim.x + threadIdx.x;
  if (i < E) atomicAdd(&deg[row[i]], 1.f);
}
__global__ void k_dinv(float* __restrict__ d, long N) {
  long i = (long)blockIdx.x * blockDim.x + threadIdx.x;
  if (i < N) d[i] = rsqrtf(d[i] + 1.f);
}
// per-column sum / sumsq of relu(h), h is [M,64]
__global__ void k_colsums(const float* __restrict__ h, long M, float* __restrict__ stats) {
  long gid = (long)blockIdx.x * blockDim.x + threadIdx.x;
  int c = (int)(gid & 63);
  long r = gid >> 6;
  long stride = ((long)gridDim.x * blockDim.x) >> 6;
  float s = 0.f, q = 0.f;
  for (; r < M; r += stride) { float v = fmaxf(h[r*64 + c], 0.f); s += v; q += v * v; }
  atomicAdd(&stats[c], s); atomicAdd(&stats[64 + c], q);
}
// BN(train, biased var) folded to per-feature affine: tsc=isd*g, tsh=bt-mu*tsc
__global__ void k_finstats(const float* __restrict__ stats, const float* __restrict__ g,
                           const float* __restrict__ bt, float invn,
                           float* __restrict__ tsc, float* __restrict__ tsh) {
  int c = threadIdx.x;
  float mu  = stats[c] * invn;
  float var = stats[64 + c] * invn - mu * mu;
  float isd = rsqrtf(var + 1e-5f);
  float a = isd * g[c];
  tsc[c] = a; tsh[c] = bt[c] - mu * a;
}

extern "C" void kernel_launch(void* const* d_in, const int* in_sizes, int n_in,
                              void* d_out, int out_size, void* d_ws, size_t ws_size,
                              hipStream_t stream) {
  (void)n_in; (void)out_size; (void)ws_size;
  const float* x   = (const float*)d_in[0];
  const float* EA  = (const float*)d_in[1];
  const int*   ei  = (const int*)  d_in[2];
  const float* W1l = (const float*)d_in[3];  const float* b1l = (const float*)d_in[4];
  const float* W1e = (const float*)d_in[5];  const float* b1e = (const float*)d_in[6];
  const float* g1  = (const float*)d_in[7];  const float* bt1 = (const float*)d_in[8];
  const float* W2l = (const float*)d_in[9];  const float* b2l = (const float*)d_in[10];
  const float* W2e = (const float*)d_in[11]; const float* b2e = (const float*)d_in[12];
  const float* g2  = (const float*)d_in[13]; const float* bt2 = (const float*)d_in[14];
  const float* W3l = (const float*)d_in[15]; const float* b3l = (const float*)d_in[16];
  const float* W3e = (const float*)d_in[17]; const float* b3e = (const float*)d_in[18];
  const float* Wu  = (const float*)d_in[19]; const float* bu  = (const float*)d_in[20];
  const float* g3  = (const float*)d_in[21]; const float* bt3 = (const float*)d_in[22];
  const float* ge3 = (const float*)d_in[23]; const float* bte3= (const float*)d_in[24];
  const float* Wn  = (const float*)d_in[25]; const float* bnv = (const float*)d_in[26];
  const float* Wel = (const float*)d_in[27]; const float* bel = (const float*)d_in[28];

  const long H  = in_sizes[4];            // 64
  const long IN = in_sizes[3] / H;        // 128
  const long N  = in_sizes[0] / IN;       // 100000
  const long EF = in_sizes[5] / H;        // 64
  const long E  = in_sizes[1] / EF;       // 1000000

  float* ws    = (float*)d_ws;
  float* dinv  = ws;                      // N floats
  float* xl    = ws + N;                  // N*64
  float* hacc  = xl + N * 64;             // N*64
  float* stats = hacc + N * 64;           // 128
  float* tsc   = stats + 128;             // 64
  float* tsh   = tsc + 64;                // 64
  // fragment-packed f16 weights (32B aligned: all prior sections are 32B multiples)
  _Float16* W1lp = (_Float16*)(tsh + 64);         // (IN/32)*2048 halves
  _Float16* W1ep = W1lp + (IN >> 5) * 2048;       // 4096 halves each below (K=64)
  _Float16* W2lp = W1ep + 4096;
  _Float16* W2ep = W2lp + 4096;
  _Float16* W3lp = W2ep + 4096;
  _Float16* W3ep = W3lp + 4096;
  _Float16* Wup  = W3ep + 4096;
  _Float16* Wnp  = Wup  + 4096;
  _Float16* Welp = Wnp  + 4096;

  auto cdiv = [](long a, long b) { return (unsigned)((a + b - 1) / b); };
  dim3 b32(32), b256(256);
  unsigned tN = cdiv(N, 16), tE = cdiv(E, 16);

  // one-time weight packs (deterministic; rerun every call)
  k_packW<<<cdiv((IN>>5)*2048,256), b256, 0, stream>>>(W1l, (int)IN, W1lp);
  k_packW<<<16, b256, 0, stream>>>(W1e, 64, W1ep);
  k_packW<<<16, b256, 0, stream>>>(W2l, 64, W2lp);
  k_packW<<<16, b256, 0, stream>>>(W2e, 64, W2ep);
  k_packW<<<16, b256, 0, stream>>>(W3l, 64, W3lp);
  k_packW<<<16, b256, 0, stream>>>(W3e, 64, W3ep);
  k_packW<<<16, b256, 0, stream>>>(Wu,  64, Wup);
  k_packW<<<16, b256, 0, stream>>>(Wn,  64, Wnp);
  k_packW<<<16, b256, 0, stream>>>(Wel, 64, Welp);

  // degree -> dinv
  k_zero<<<cdiv(N,256), b256, 0, stream>>>(dinv, N);
  k_deg <<<cdiv(E,256), b256, 0, stream>>>(ei, dinv, E);
  k_dinv<<<cdiv(N,256), b256, 0, stream>>>(dinv, N);

  // ---- layer 1 ----
  k_gemm<false><<<tN, b32, 0, stream>>>(x, (int)IN, N, W1lp, b1l, nullptr, nullptr, xl);
  k_zero<<<cdiv(N*64,256), b256, 0, stream>>>(hacc, N*64);
  k_msg <<<tE, b32, 0, stream>>>(EA, W1ep, b1e, ei, E, xl, dinv, hacc);
  k_zero<<<1, 128, 0, stream>>>(stats, 128);
  k_colsums<<<256, b256, 0, stream>>>(hacc, N, stats);
  k_finstats<<<1, 64, 0, stream>>>(stats, g1, bt1, 1.0f/(float)N, tsc, tsh);

  // ---- layer 2 (BN fused into A-load, branch-free) ----
  k_gemm<true><<<tN, b32, 0, stream>>>(hacc, 64, N, W2lp, b2l, tsc, tsh, xl);
  k_zero<<<cdiv(N*64,256), b256, 0, stream>>>(hacc, N*64);
  k_msg <<<tE, b32, 0, stream>>>(EA, W2ep, b2e, ei, E, xl, dinv, hacc);
  k_zero<<<1, 128, 0, stream>>>(stats, 128);
  k_colsums<<<256, b256, 0, stream>>>(hacc, N, stats);
  k_finstats<<<1, 64, 0, stream>>>(stats, g2, bt2, 1.0f/(float)N, tsc, tsh);

  // ---- layer 3 ----
  k_gemm<true><<<tN, b32, 0, stream>>>(hacc, 64, N, W3lp, b3l, tsc, tsh, xl);
  k_zero<<<cdiv(N*64,256), b256, 0, stream>>>(hacc, N*64);
  k_msg <<<tE, b32, 0, stream>>>(EA, W3ep, b3e, ei, E, xl, dinv, hacc);  // hacc = h3 (pre-BN)

  // ---- node_out = bn(relu(h3)) @ Wn + bn ----
  k_zero<<<1, 128, 0, stream>>>(stats, 128);
  k_colsums<<<256, b256, 0, stream>>>(hacc, N, stats);
  k_finstats<<<1, 64, 0, stream>>>(stats, g3, bt3, 1.0f/(float)N, tsc, tsh);
  float* node_out = (float*)d_out;
  k_gemm<true><<<tN, b32, 0, stream>>>(hacc, 64, N, Wnp, bnv, tsc, tsh, node_out);

  // ---- edge_out = bn(relu(new_e)) @ We_lin + be_lin  (new_e recomputed, never stored) ----
  k_zero<<<1, 128, 0, stream>>>(stats, 128);
  k_newe_stats<<<tE, b32, 0, stream>>>(EA, W3ep, b3e, Wup, bu, ei, E, hacc, stats);
  k_finstats<<<1, 64, 0, stream>>>(stats, ge3, bte3, 1.0f/(float)E, tsc, tsh);
  k_edge_out<<<tE, b32, 0, stream>>>(EA, W3ep, b3e, Wup, bu, ei, E, hacc,
                                     tsc, tsh, Welp, bel, node_out + N*64);
}
